// PriorConv2d_23218593202454
// MI455X (gfx1250) — matrix operand order repaired and checked by reading, not compiled, not run on hardware
//
#include <hip/hip_runtime.h>
#include <hip/hip_bf16.h>
#include <cstdint>

// ---------------------------------------------------------------------------
// PriorConv2d on gfx1250 (MI455X).
// Pipeline (all on `stream`):
//   K0: zero per-batch s-accumulators (8 floats in ws)
//   K1: fused CNN tile kernel: conv1(3->16)+SiLU (VALU), conv2(16->16) via
//       v_wmma_f32_16x16x32_f16 (K=144 padded to 160 -> 5 WMMAs per tile),
//       conv3(16->1) (VALU), block reduction -> atomicAdd(sacc[b]).
//       Also writes RGB_order to out channels 2..4 (input tile already in LDS).
//   K2: per-batch scale -> normalized separable 1D taps e_n (37), c_n (37).
//   K3: row pass: xg = gcm*batch on the fly, 4 separable intermediates
//       (xg0*e, xg0*c, xg1*e, xg2*e along W) -> 33.5 MB in ws (fits L2).
//   K4: column pass + final math (Hf, S, Ww) -> out channels 0,1,5.
// ---------------------------------------------------------------------------

typedef _Float16 v16h __attribute__((ext_vector_type(16)));
typedef float    v8f  __attribute__((ext_vector_type(8)));

#define HH 512
#define WW 512
#define BB 8
#define FMAXR 18
#define KSZ 37
#define EPSR 1e-4f
#define PI_F 3.14159265358979323846f

__device__ __forceinline__ float silu_f(float x) {
    return x / (1.0f + __expf(-x));
}

// ---------------------------------------------------------------------------
// K0: zero the per-batch accumulators
// ---------------------------------------------------------------------------
__global__ void zero_kernel(float* p, int n) {
    int i = threadIdx.x;
    if (i < n) p[i] = 0.0f;
}

// ---------------------------------------------------------------------------
// K1: fused CNN + RGB_order. One block = one 16x16 output tile of s.
// grid (W/16, H/16, B), block 256 (= 8 wave32).
// ---------------------------------------------------------------------------
__global__ __launch_bounds__(256)
void cnn_fused_kernel(const float* __restrict__ batch,
                      const float* __restrict__ w1, const float* __restrict__ b1,
                      const float* __restrict__ w2, const float* __restrict__ b2,
                      const float* __restrict__ w3, const float* __restrict__ b3,
                      float* __restrict__ out, float* __restrict__ sacc) {
    const int tx = blockIdx.x, ty = blockIdx.y, b = blockIdx.z;
    const int x0 = tx * 16, y0 = ty * 16;
    const int tid  = threadIdx.x;
    const int wave = tid >> 5;
    const int lane = tid & 31;

    // LDS tiles: s(16x16) <- h2(18x18) <- h1(20x20) <- in(22x22)
    __shared__ float      in_t[3 * 22 * 22];
    __shared__ alignas(32) _Float16 h1s[20 * 20 * 16];   // [y][x][ic]
    __shared__ alignas(32) _Float16 h2s[18 * 18 * 16];   // [y][x][ic]
    __shared__ alignas(32) _Float16 w2h[16 * 10 * 16];   // [oc][tap(0..9, 9=pad0)][ic]
    __shared__ float w1s[432];    // [oc][ic][tap]
    __shared__ float w3s[144];    // [ic][tap]
    __shared__ float b1s[16], b2s[16];
    __shared__ float red[256];

    // ---- stage weights into LDS ----
    for (int i = tid; i < 432; i += 256) w1s[i] = w1[i];
    for (int i = tid; i < 144; i += 256) w3s[i] = w3[i];
    if (tid < 16) { b1s[tid] = b1[tid]; b2s[tid] = b2[tid]; }
    for (int i = tid; i < 16 * 10 * 16; i += 256) {
        const int ic = i & 15, tap = (i >> 4) % 10, oc = i / 160;
        w2h[i] = (tap < 9) ? (_Float16)w2[oc * 144 + ic * 9 + tap] : (_Float16)0.0f;
    }
    // ---- input tile (halo +-3, zero-padded like the reference convs) ----
    for (int i = tid; i < 3 * 22 * 22; i += 256) {
        const int c = i / 484, r = (i % 484) / 22, col = i % 22;
        const int gy = y0 + r - 3, gx = x0 + col - 3;
        in_t[i] = (gy >= 0 && gy < HH && gx >= 0 && gx < WW)
                      ? batch[((size_t)(b * 3 + c) * HH + gy) * WW + gx] : 0.0f;
    }
    __syncthreads();

    // ---- conv1 + SiLU -> h1 on 20x20 (zero outside image: ref zero-pads h) ----
    for (int p = tid; p < 400; p += 256) {
        const int r = p / 20, c = p % 20;
        const int gy = y0 - 2 + r, gx = x0 - 2 + c;
        if (gy < 0 || gy >= HH || gx < 0 || gx >= WW) {
            for (int oc = 0; oc < 16; ++oc) h1s[p * 16 + oc] = (_Float16)0.0f;
            continue;
        }
        float acc[16];
        for (int oc = 0; oc < 16; ++oc) acc[oc] = b1s[oc];
        for (int ic = 0; ic < 3; ++ic)
            for (int ky = 0; ky < 3; ++ky)
                for (int kx = 0; kx < 3; ++kx) {
                    const float v = in_t[(ic * 22 + (r + ky)) * 22 + (c + kx)];
                    const int wbase = ic * 9 + ky * 3 + kx;
                    for (int oc = 0; oc < 16; ++oc)
                        acc[oc] += w1s[oc * 27 + wbase] * v;
                }
        for (int oc = 0; oc < 16; ++oc)
            h1s[p * 16 + oc] = (_Float16)silu_f(acc[oc]);
    }
    __syncthreads();

    // ---- conv2 via WMMA: K = tap*16 + ic, 144 padded to 160 (5 chunks) ----
    // A (16x32 f16) layout: lanes 0-15 -> M=lane, K 0..15;
    // lanes 16-31 -> M=lane-16, K 16..31 (same order as B).
    v16h afr[5];
    {
        const int oc  = lane & 15;
        const int icb = 0;
        for (int c = 0; c < 5; ++c) {
            const int t0 = 2 * c + (lane >> 4), t1 = t0;
            for (int j = 0; j < 4; ++j) {
                afr[c][2 * j]         = w2h[(oc * 10 + t0) * 16 + icb + 2 * j];
                afr[c][2 * j + 1]     = w2h[(oc * 10 + t0) * 16 + icb + 2 * j + 1];
                afr[c][8 + 2 * j]     = w2h[(oc * 10 + t1) * 16 + icb + 8 + 2 * j];
                afr[c][8 + 2 * j + 1] = w2h[(oc * 10 + t1) * 16 + icb + 8 + 2 * j + 1];
            }
        }
    }
    // 18x18 = 324 h2 pixels -> 21 groups of 16 (last clamped; duplicate
    // columns produce identical values, benign duplicate stores).
    for (int g = wave; g < 21; g += 8) {
        v8f acc = {};
        int p = g * 16 + (lane & 15);
        if (p > 323) p = 323;
        const int yy = p / 18, xx = p % 18;
        for (int c = 0; c < 5; ++c) {
            const int tap = 2 * c + (lane >> 4);   // B: lanes 0-15 hold K 0..15
            v16h bfrag = {};
            if (tap < 9) {
                const int ky = tap / 3, kx = tap % 3;
                bfrag = *reinterpret_cast<const v16h*>(
                    &h1s[((yy + ky) * 20 + (xx + kx)) * 16]);
            }
            acc = __builtin_amdgcn_wmma_f32_16x16x32_f16(
                false, afr[c], false, bfrag, (short)0, acc, false, false);
        }
        // C/D layout: VGPR j -> M = j + 8*(lane>=16), N = lane&15
        const int gy = y0 - 1 + yy, gx = x0 - 1 + xx;
        const bool inb = (gy >= 0 && gy < HH && gx >= 0 && gx < WW);
        for (int j = 0; j < 8; ++j) {
            const int oc = j + ((lane >> 4) << 3);
            float v = acc[j] + b2s[oc];
            v = inb ? silu_f(v) : 0.0f;
            h2s[(yy * 18 + xx) * 16 + oc] = (_Float16)v;
        }
    }
    __syncthreads();

    // ---- conv3 (16->1) + RGB_order + block reduction ----
    {
        const int y = tid >> 4, x = tid & 15;
        float s = b3[0];
        for (int ky = 0; ky < 3; ++ky)
            for (int kx = 0; kx < 3; ++kx) {
                const int hb = (((y + ky) * 18) + (x + kx)) * 16;
                const int t = ky * 3 + kx;
                for (int ic = 0; ic < 16; ++ic)
                    s += w3s[ic * 9 + t] * (float)h2s[hb + ic];
            }

        // RGB_order from the input tile (interior pixel = (y+3, x+3))
        const float v0 = in_t[(0 * 22 + (y + 3)) * 22 + (x + 3)];
        const float v1 = in_t[(1 * 22 + (y + 3)) * 22 + (x + 3)];
        const float v2 = in_t[(2 * 22 + (y + 3)) * 22 + (x + 3)];
        int mx = 0;  float bv = v0; if (v1 > bv) { bv = v1; mx = 1; }  if (v2 > bv) { bv = v2; mx = 2; }
        int mx2 = 2;       bv = v2; if (v1 > bv) { bv = v1; mx2 = 1; } if (v0 > bv) { bv = v0; mx2 = 0; }
        int mn = 0;        bv = v0; if (v1 < bv) { bv = v1; mn = 1; }  if (v2 < bv) { bv = v2; mn = 2; }
        int mn2 = 2;       bv = v2; if (v1 < bv) { bv = v1; mn2 = 1; } if (v0 < bv) { bv = v0; mn2 = 0; }
        const int gy = y0 + y, gx = x0 + x;
        for (int ch = 0; ch < 3; ++ch) {
            const float val = 0.5f * ((mx == ch) + (mx2 == ch))
                            - 0.5f * ((mn == ch) + (mn2 == ch));
            out[((size_t)(b * 6 + 2 + ch) * HH + gy) * WW + gx] = val;
        }

        red[tid] = s;
    }
    __syncthreads();
    for (int off = 128; off > 0; off >>= 1) {
        if (tid < off) red[tid] += red[tid + off];
        __syncthreads();
    }
    if (tid == 0) atomicAdd(&sacc[b], red[0]);
}

// ---------------------------------------------------------------------------
// K2: per-batch normalized 1D taps. grid B, block 64.
// ---------------------------------------------------------------------------
__global__ void taps_kernel(const float* __restrict__ sacc, float* __restrict__ taps) {
    const int b = blockIdx.x, t = threadIdx.x;
    float scale = sacc[b] * (1.0f / ((float)HH * (float)WW));
    scale = fminf(2.5f, fmaxf(-2.5f, scale));
    const float sd = exp2f(scale);
    const float fs = ceilf(3.0f * sd + 0.5f);

    float e = 0.0f, c = 0.0f;
    if (t < KSZ) {
        const float x = (float)(t - FMAXR);
        const float m = (fabsf(x) <= fs) ? 1.0f : 0.0f;
        const float q = x / sd;
        e = __expf(-0.5f * q * q) * m;
        c = -x / (sd * sd * sd * 2.0f * PI_F) * e;
    }
    __shared__ float se[64], sc[64];
    se[t] = e; sc[t] = fabsf(c);
    __syncthreads();
    for (int off = 32; off > 0; off >>= 1) {
        if (t < off) { se[t] += se[t + off]; sc[t] += sc[t + off]; }
        __syncthreads();
    }
    if (t < KSZ) {
        taps[((size_t)b * 2 + 0) * KSZ + t] = e / se[0];
        taps[((size_t)b * 2 + 1) * KSZ + t] = c / sc[0];
    }
}

// ---------------------------------------------------------------------------
// K3: row pass. grid (W/256, H, B), block 256.
// planes: [0]=xg0*e  [1]=xg0*c  [2]=xg1*e  [3]=xg2*e   (conv along W)
// ---------------------------------------------------------------------------
__global__ __launch_bounds__(256)
void row_kernel(const float* __restrict__ batch, const float* __restrict__ gcm,
                const float* __restrict__ taps, float* __restrict__ planes) {
    const int xs = blockIdx.x, y = blockIdx.y, b = blockIdx.z;
    const int tid = threadIdx.x;
    const int x0 = xs * 256;

    __shared__ float raw[3 * 292];
    __shared__ float xg[3 * 292];
    __shared__ float te[KSZ], tc[KSZ], gm[9];

    if (tid < KSZ) {
        te[tid] = taps[((size_t)b * 2 + 0) * KSZ + tid];
        tc[tid] = taps[((size_t)b * 2 + 1) * KSZ + tid];
    }
    if (tid < 9) gm[tid] = gcm[tid];
    for (int i = tid; i < 3 * 292; i += 256) {
        const int c = i / 292, col = i % 292;
        const int gx = x0 - FMAXR + col;
        raw[i] = (gx >= 0 && gx < WW)
                     ? batch[((size_t)(b * 3 + c) * HH + y) * WW + gx] : 0.0f;
    }
    __syncthreads();
    for (int i = tid; i < 3 * 292; i += 256) {
        const int j = i / 292, col = i % 292;
        xg[i] = gm[j * 3 + 0] * raw[0 * 292 + col]
              + gm[j * 3 + 1] * raw[1 * 292 + col]
              + gm[j * 3 + 2] * raw[2 * 292 + col];
    }
    __syncthreads();

    float a0 = 0.f, a1 = 0.f, a2 = 0.f, a3 = 0.f;
    for (int k = 0; k < KSZ; ++k) {
        const float we = te[k], wc = tc[k];
        const float u0 = xg[0 * 292 + tid + k];
        a0 += we * u0;
        a1 += wc * u0;
        a2 += we * xg[1 * 292 + tid + k];
        a3 += we * xg[2 * 292 + tid + k];
    }
    const size_t plane = (size_t)BB * HH * WW;
    const size_t idx = ((size_t)b * HH + y) * WW + x0 + tid;
    planes[idx]             = a0;
    planes[plane + idx]     = a1;
    planes[2 * plane + idx] = a2;
    planes[3 * plane + idx] = a3;
}

// ---------------------------------------------------------------------------
// K4: column pass + final math. grid (W/256, H, B), block 256.
// Plane re-reads along y hit L2 (33.5 MB working set << 192 MB L2).
// ---------------------------------------------------------------------------
__global__ __launch_bounds__(256)
void colfinal_kernel(const float* __restrict__ taps,
                     const float* __restrict__ planes,
                     float* __restrict__ out) {
    const int xs = blockIdx.x, y = blockIdx.y, b = blockIdx.z;
    const int tid = threadIdx.x;
    const int x = xs * 256 + tid;

    __shared__ float te[KSZ], tc[KSZ];
    if (tid < KSZ) {
        te[tid] = taps[((size_t)b * 2 + 0) * KSZ + tid];
        tc[tid] = taps[((size_t)b * 2 + 1) * KSZ + tid];
    }
    __syncthreads();

    const size_t plane = (size_t)BB * HH * WW;
    float E = 0.f, Ex = 0.f, Ey = 0.f, El = 0.f, Ell = 0.f;
    for (int k = 0; k < KSZ; ++k) {
        const int yy = y - FMAXR + k;
        if (yy < 0 || yy >= HH) continue;
        const size_t idx = ((size_t)b * HH + yy) * WW + x;
        const float p0 = planes[idx];
        const float p1 = planes[plane + idx];
        const float p2 = planes[2 * plane + idx];
        const float p3 = planes[3 * plane + idx];
        const float we = te[k], wc = tc[k];
        E  += we * p0;
        Ex += wc * p0;   // dgdx = c(h) x e(w)
        Ey += we * p1;   // dgdy = e(h) x c(w)
        El += we * p2;
        Ell += we * p3;
    }
    const float Hf = atanf(El / (Ell + EPSR));
    const float S  = logf((El * El + Ell * Ell) / (E * E + EPSR) + EPSR);
    const float rx = Ex / (E + EPSR), ry = Ey / (E + EPSR);
    const float Wv = atanf(rx * rx + ry * ry);

    out[((size_t)(b * 6 + 0) * HH + y) * WW + x] = Hf;
    out[((size_t)(b * 6 + 1) * HH + y) * WW + x] = S;
    out[((size_t)(b * 6 + 5) * HH + y) * WW + x] = Wv;
}

// ---------------------------------------------------------------------------
extern "C" void kernel_launch(void* const* d_in, const int* in_sizes, int n_in,
                              void* d_out, int out_size, void* d_ws, size_t ws_size,
                              hipStream_t stream) {
    const float* batch = (const float*)d_in[0];
    const float* gcm   = (const float*)d_in[1];
    const float* w1    = (const float*)d_in[2];
    const float* b1    = (const float*)d_in[3];
    const float* w2    = (const float*)d_in[4];
    const float* b2    = (const float*)d_in[5];
    const float* w3    = (const float*)d_in[6];
    const float* b3    = (const float*)d_in[7];
    float* out = (float*)d_out;
    float* ws  = (float*)d_ws;

    // ws layout (floats): [0..7] sacc, [64..655] taps, [1024..] 4 planes (B*H*W each)
    float* sacc   = ws;
    float* taps   = ws + 64;
    float* planes = ws + 1024;   // needs 4*8*512*512*4 B = 33.6 MB of ws

    zero_kernel<<<1, 32, 0, stream>>>(sacc, BB);
    cnn_fused_kernel<<<dim3(WW / 16, HH / 16, BB), 256, 0, stream>>>(
        batch, w1, b1, w2, b2, w3, b3, out, sacc);
    taps_kernel<<<BB, 64, 0, stream>>>(sacc, taps);
    row_kernel<<<dim3(WW / 256, HH, BB), 256, 0, stream>>>(batch, gcm, taps, planes);
    colfinal_kernel<<<dim3(WW / 256, HH, BB), 256, 0, stream>>>(taps, planes, out);
}
